// TutelMOE_44848048505006
// MI455X (gfx1250) — compile-verified, matrix operand order checked by reading, not run on hardware
//
#include <hip/hip_runtime.h>
#include <hip/hip_bf16.h>
#include <math.h>

// ---------------- problem constants ----------------
#define S_TOK   8192      // B*L = 4*2048
#define DMODEL  2048
#define HIDDEN  8192
#define NEXP    8
#define TOPK    2
#define CAP     2048      // (TOPK*S)/NEXP

// ---------------- WMMA types ----------------
typedef __attribute__((ext_vector_type(16))) __bf16 v16bf;
typedef __attribute__((ext_vector_type(8)))  __bf16 bf16x8;
typedef __attribute__((ext_vector_type(2)))  __bf16 bf16x2;
typedef __attribute__((ext_vector_type(8)))  float  v8f;

// async-copy pointer types (from hipcc diagnostic: param is v4i32 in AS1/AS3)
typedef int v4i __attribute__((vector_size(16)));
typedef __attribute__((address_space(1))) v4i as1_v4i;
typedef __attribute__((address_space(3))) v4i as3_v4i;

// ---------------- GEMM tile config ----------------
#define BM 128
#define BN 128
#define BK 32
#define ASTR 40   // padded LDS strides (bf16 elems): 80B rows -> conflict-free 16-row frags
#define BSTR 40

// CDNA5 async global->LDS path (ASYNCcnt), with portable fallback
#if __has_builtin(__builtin_amdgcn_global_load_async_to_lds_b128) && \
    __has_builtin(__builtin_amdgcn_s_wait_asynccnt)
#define USE_ASYNC_COPY 1
#else
#define USE_ASYNC_COPY 0
#endif

// =====================================================================
// Kernel 1: gating — scores = softmax(x @ wg), top-2, normalized gates
// one wave (32 lanes) per token
// =====================================================================
__global__ __launch_bounds__(256) void moe_gate_kernel(
    const float* __restrict__ x, const float* __restrict__ wg,
    int* __restrict__ idxb, float* __restrict__ gateb)
{
    const int lane  = threadIdx.x & 31;
    const int wid   = threadIdx.x >> 5;
    const int token = blockIdx.x * 8 + wid;
    if (token >= S_TOK) return;

    float acc[NEXP];
#pragma unroll
    for (int e = 0; e < NEXP; ++e) acc[e] = 0.f;

    const float* xr = x + (size_t)token * DMODEL;
    for (int d = lane; d < DMODEL; d += 32) {
        const float xv = xr[d];
        const float* wr = wg + (size_t)d * NEXP;
#pragma unroll
        for (int e = 0; e < NEXP; ++e) acc[e] += xv * wr[e];
    }
    // wave32 reduction
#pragma unroll
    for (int off = 16; off > 0; off >>= 1) {
#pragma unroll
        for (int e = 0; e < NEXP; ++e) acc[e] += __shfl_xor(acc[e], off, 32);
    }
    if (lane == 0) {
        float mx = acc[0];
#pragma unroll
        for (int e = 1; e < NEXP; ++e) mx = fmaxf(mx, acc[e]);
        float p[NEXP], sum = 0.f;
#pragma unroll
        for (int e = 0; e < NEXP; ++e) { p[e] = __expf(acc[e] - mx); sum += p[e]; }
        const float inv = 1.f / sum;
#pragma unroll
        for (int e = 0; e < NEXP; ++e) p[e] *= inv;
        // top-2, lowest index wins ties (matches jax top_k)
        int i0 = 0;
        for (int e = 1; e < NEXP; ++e) if (p[e] > p[i0]) i0 = e;
        int i1 = (i0 == 0) ? 1 : 0;
        for (int e = 0; e < NEXP; ++e) if (e != i0 && p[e] > p[i1]) i1 = e;
        const float g0 = p[i0], g1 = p[i1];
        const float den = fmaxf(g0 + g1, 1e-9f);
        idxb[token * 2 + 0]  = i0;
        idxb[token * 2 + 1]  = i1;
        gateb[token * 2 + 0] = g0 / den;
        gateb[token * 2 + 1] = g1 / den;
    }
}

// =====================================================================
// Kernel 2: deterministic capacity scan (single workgroup)
// =====================================================================
__global__ __launch_bounds__(256) void moe_scan_kernel(
    const int* __restrict__ idxb, const float* __restrict__ gateb,
    int* __restrict__ slot_token, float* __restrict__ slot_gate)
{
    __shared__ int cnt[256][NEXP];
    __shared__ int totals[NEXP];
    const int tid = threadIdx.x;

    for (int i = tid; i < NEXP * CAP; i += 256) {
        slot_token[i] = -1;
        slot_gate[i]  = 0.f;
    }
    __syncthreads();

    int carry[NEXP];
#pragma unroll
    for (int e = 0; e < NEXP; ++e) carry[e] = 0;

    const int t0 = tid * (S_TOK / 256);
    const int t1 = t0 + (S_TOK / 256);

    for (int s = 0; s < TOPK; ++s) {
        int local[NEXP];
#pragma unroll
        for (int e = 0; e < NEXP; ++e) local[e] = 0;
        for (int t = t0; t < t1; ++t) local[idxb[t * 2 + s]]++;
#pragma unroll
        for (int e = 0; e < NEXP; ++e) cnt[tid][e] = local[e];
        __syncthreads();
        if (tid == 0) {
            int run[NEXP];
#pragma unroll
            for (int e = 0; e < NEXP; ++e) run[e] = 0;
            for (int j = 0; j < 256; ++j)
                for (int e = 0; e < NEXP; ++e) {
                    const int tmp = cnt[j][e];
                    cnt[j][e] = run[e];
                    run[e] += tmp;
                }
            for (int e = 0; e < NEXP; ++e) totals[e] = run[e];
        }
        __syncthreads();
        int base[NEXP];
#pragma unroll
        for (int e = 0; e < NEXP; ++e) base[e] = cnt[tid][e] + carry[e];
        for (int t = t0; t < t1; ++t) {
            const int e = idxb[t * 2 + s];
            const int p = base[e]++;
            if (p < CAP) {
                slot_token[e * CAP + p] = t;
                slot_gate[e * CAP + p]  = gateb[t * 2 + s];
            }
        }
        __syncthreads();
#pragma unroll
        for (int e = 0; e < NEXP; ++e) carry[e] += totals[e];
        __syncthreads();
    }
}

// =====================================================================
// Kernel 3: dispatch — disp[e][pos][:] = bf16(x[token]), zeros elsewhere
// =====================================================================
__global__ __launch_bounds__(256) void moe_dispatch_kernel(
    const float* __restrict__ x, const int* __restrict__ slot_token,
    __bf16* __restrict__ disp)
{
    const int slot = blockIdx.x;
    const int tok  = slot_token[slot];
    __bf16* dst = disp + (size_t)slot * DMODEL;
    if (tok >= 0) {
        const float* src = x + (size_t)tok * DMODEL;
        for (int i = threadIdx.x; i < DMODEL; i += 256) dst[i] = (__bf16)src[i];
    } else {
        for (int i = threadIdx.x; i < DMODEL; i += 256) dst[i] = (__bf16)0.0f;
    }
}

__global__ __launch_bounds__(256) void moe_zero_kernel(float* __restrict__ p, int n)
{
    const int i = blockIdx.x * 256 + threadIdx.x;
    if (i < n) p[i] = 0.f;
}

// =====================================================================
// helpers
// =====================================================================
__device__ __forceinline__ unsigned pack2_bf16(float lo, float hi)
{
#if __has_builtin(__builtin_amdgcn_cvt_pk_bf16_f32)
    auto p = __builtin_amdgcn_cvt_pk_bf16_f32(lo, hi);   // single v_cvt_pk_bf16_f32
    return __builtin_bit_cast(unsigned, p);
#else
    bf16x2 p;
    p[0] = (__bf16)lo;
    p[1] = (__bf16)hi;
    return __builtin_bit_cast(unsigned, p);
#endif
}

__device__ __forceinline__ float gelu_exact(float v)
{
    return 0.5f * v * (1.0f + erff(v * 0.70710678118654752f));
}

// =====================================================================
// WMMA bf16 mainloop, double-buffered LDS, 1 barrier per K-step.
// C_tile(BM x BN) += A(bf16)[MxK] @ bf16(B fp32)[KxN]
// 8 waves in a 4x2 grid; per-wave tile 32x64 -> 8 WMMAs per K-step.
// A tile: bf16, moved global->LDS with GLOBAL_LOAD_ASYNC_TO_LDS (ASYNCcnt)
//         when the toolchain exposes it (register-prefetch fallback else).
// B tile: fp32 weights, converted with packed v_cvt_pk_bf16_f32 and stored
//         K-transposed so all fragment reads are contiguous 16B ds_load_b128.
// =====================================================================
__device__ __forceinline__ void gemm_mainloop(
    const __bf16* __restrict__ A, int lda,
    const float*  __restrict__ B, int ldb,
    int K, int gm0, int gn0, v8f (&acc)[2][4])
{
    __shared__ __align__(16) __bf16 sA[2][BM * ASTR];
    __shared__ __align__(16) __bf16 sB[2][BN * BSTR];

    const int tid  = threadIdx.x;
    const int lane = tid & 31;
    const int w    = tid >> 5;
    const int wm   = w & 3;            // 4 wave rows -> 32 rows each
    const int wn   = w >> 2;           // 2 wave cols -> 64 cols each
    const int l15  = lane & 15;
    const int klo  = (lane >> 4) * 8;  // fragment K sub-offset per lane half

    float4 b0R[2], b1R[2];

    auto load_B_regs = [&](int k0) {
#pragma unroll
        for (int c = 0; c < 2; ++c) {
            const int q  = tid * 2 + c;             // 512 items: 16 k-pairs x 32 n-groups
            const int kp = q >> 5;                  // 0..15 -> k = 2*kp
            const int n  = (q & 31) * 4;            // 0..124
            const float* gp = B + (size_t)(k0 + kp * 2) * ldb + (gn0 + n);
            b0R[c] = *(const float4*)gp;
            b1R[c] = *(const float4*)(gp + ldb);
        }
    };

    auto store_B_tile = [&](int cur) {
#pragma unroll
        for (int c = 0; c < 2; ++c) {
            const int q  = tid * 2 + c;
            const int k  = (q >> 5) * 2;
            const int n  = (q & 31) * 4;
#pragma unroll
            for (int j = 0; j < 4; ++j) {
                const float lo = (&b0R[c].x)[j];    // B[k  ][n+j]
                const float hi = (&b1R[c].x)[j];    // B[k+1][n+j]
                *(unsigned*)(&sB[cur][(n + j) * BSTR + k]) = pack2_bf16(lo, hi);
            }
        }
    };

#if USE_ASYNC_COPY
    auto async_load_A = [&](int k0, int buf) {
#pragma unroll
        for (int c = 0; c < 2; ++c) {
            const int chunk = tid * 2 + c;          // 512 chunks of 8 bf16 (16B)
            const int row   = chunk >> 2;
            const int kc    = (chunk & 3) * 8;
            const __bf16* g = A + (size_t)(gm0 + row) * lda + (k0 + kc);
            __builtin_amdgcn_global_load_async_to_lds_b128(
                (as1_v4i*)g,
                (as3_v4i*)&sA[buf][row * ASTR + kc],
                0, 0);
        }
    };
#else
    bf16x8 aR[2];
    auto load_A_regs = [&](int k0) {
#pragma unroll
        for (int c = 0; c < 2; ++c) {
            const int chunk = tid * 2 + c;
            const int row   = chunk >> 2;
            const int kc    = (chunk & 3) * 8;
            aR[c] = *(const bf16x8*)(A + (size_t)(gm0 + row) * lda + (k0 + kc));
        }
    };
    auto store_A_tile = [&](int cur) {
#pragma unroll
        for (int c = 0; c < 2; ++c) {
            const int chunk = tid * 2 + c;
            const int row   = chunk >> 2;
            const int kc    = (chunk & 3) * 8;
            *(bf16x8*)(&sA[cur][row * ASTR + kc]) = aR[c];
        }
    };
#endif

#if USE_ASYNC_COPY
    async_load_A(0, 0);
    load_B_regs(0);
#else
    load_A_regs(0);
    load_B_regs(0);
#endif

    int cur = 0;
    for (int k0 = 0; k0 < K; k0 += BK) {
#if USE_ASYNC_COPY
        store_B_tile(cur);
        __builtin_amdgcn_s_wait_asynccnt(0);        // A tile `cur` resident in LDS
#else
        store_A_tile(cur);
        store_B_tile(cur);
#endif
        __syncthreads();
        if (k0 + BK < K) {
#if USE_ASYNC_COPY
            async_load_A(k0 + BK, cur ^ 1);         // async copy overlaps WMMA below
#else
            load_A_regs(k0 + BK);
#endif
            load_B_regs(k0 + BK);
            if (k0 + 2 * BK < K) {                  // prefetch weight stream 2 tiles ahead
                const float* pf = B + (size_t)(k0 + 2 * BK + ((tid * 2) >> 5) * 2) * ldb
                                    + (gn0 + ((tid * 2) & 31) * 4);
                __builtin_prefetch(pf, 0, 0);
            }
        }

        v16bf afrag[2], bfrag[4];
#pragma unroll
        for (int mt = 0; mt < 2; ++mt) {
            const __bf16* p = &sA[cur][(wm * 32 + mt * 16 + l15) * ASTR + klo];
            bf16x8 lo = *(const bf16x8*)p;
            bf16x8 hi = *(const bf16x8*)(p + 16);
            afrag[mt] = __builtin_shufflevector(lo, hi, 0,1,2,3,4,5,6,7,8,9,10,11,12,13,14,15);
        }
#pragma unroll
        for (int nt = 0; nt < 4; ++nt) {
            const __bf16* p = &sB[cur][(wn * 64 + nt * 16 + l15) * BSTR + klo];
            bf16x8 lo = *(const bf16x8*)p;
            bf16x8 hi = *(const bf16x8*)(p + 16);
            bfrag[nt] = __builtin_shufflevector(lo, hi, 0,1,2,3,4,5,6,7,8,9,10,11,12,13,14,15);
        }
#pragma unroll
        for (int mt = 0; mt < 2; ++mt)
#pragma unroll
            for (int nt = 0; nt < 4; ++nt)
                acc[mt][nt] = __builtin_amdgcn_wmma_f32_16x16x32_bf16(
                    false, afrag[mt], false, bfrag[nt],
                    (short)0, acc[mt][nt], false, false);
        cur ^= 1;
    }
}

// =====================================================================
// Kernel 4: per-expert fc1 + bias + GeLU(exact) -> h (bf16)
// grid (CAP/BM, HIDDEN/BN) = (16, 64)
// =====================================================================
__global__ __launch_bounds__(256) void moe_fc1_kernel(
    const __bf16* __restrict__ disp_e, const float* __restrict__ w1_e,
    const float* __restrict__ b1_e, __bf16* __restrict__ h_e)
{
    v8f acc[2][4];
#pragma unroll
    for (int mt = 0; mt < 2; ++mt)
#pragma unroll
        for (int nt = 0; nt < 4; ++nt)
#pragma unroll
            for (int i = 0; i < 8; ++i) acc[mt][nt][i] = 0.f;

    const int gm0 = blockIdx.x * BM;
    const int gn0 = blockIdx.y * BN;
    gemm_mainloop(disp_e, DMODEL, w1_e, HIDDEN, DMODEL, gm0, gn0, acc);

    const int lane = threadIdx.x & 31;
    const int w    = threadIdx.x >> 5;
    const int wm   = w & 3, wn = w >> 2;
    const int l15  = lane & 15;
    const int half = lane >> 4;
#pragma unroll
    for (int mt = 0; mt < 2; ++mt)
#pragma unroll
        for (int nt = 0; nt < 4; ++nt) {
            const int n = gn0 + wn * 64 + nt * 16 + l15;
            const float b = b1_e[n];
#pragma unroll
            for (int i = 0; i < 8; ++i) {
                const int m = gm0 + wm * 32 + mt * 16 + half * 8 + i;
                h_e[(size_t)m * HIDDEN + n] = (__bf16)gelu_exact(acc[mt][nt][i] + b);
            }
        }
}

// =====================================================================
// Kernel 5: per-expert fc2 + bias, fused combine: y[token] += g*(acc+b2)
// grid (CAP/BM, DMODEL/BN) = (16, 16). Experts serialized on stream ->
// each token row touched at most once per launch, plain RMW is safe.
// =====================================================================
__global__ __launch_bounds__(256) void moe_fc2_combine_kernel(
    const __bf16* __restrict__ h_e, const float* __restrict__ w2_e,
    const float* __restrict__ b2_e, const int* __restrict__ slot_token_e,
    const float* __restrict__ slot_gate_e, float* __restrict__ y)
{
    v8f acc[2][4];
#pragma unroll
    for (int mt = 0; mt < 2; ++mt)
#pragma unroll
        for (int nt = 0; nt < 4; ++nt)
#pragma unroll
            for (int i = 0; i < 8; ++i) acc[mt][nt][i] = 0.f;

    const int gm0 = blockIdx.x * BM;
    const int gn0 = blockIdx.y * BN;
    gemm_mainloop(h_e, HIDDEN, w2_e, DMODEL, HIDDEN, gm0, gn0, acc);

    const int lane = threadIdx.x & 31;
    const int w    = threadIdx.x >> 5;
    const int wm   = w & 3, wn = w >> 2;
    const int l15  = lane & 15;
    const int half = lane >> 4;
#pragma unroll
    for (int mt = 0; mt < 2; ++mt)
#pragma unroll
        for (int nt = 0; nt < 4; ++nt) {
            const int n = gn0 + wn * 64 + nt * 16 + l15;
            const float b = b2_e[n];
#pragma unroll
            for (int i = 0; i < 8; ++i) {
                const int m   = gm0 + wm * 32 + mt * 16 + half * 8 + i;
                const int tok = slot_token_e[m];
                if (tok >= 0) {
                    const float g = slot_gate_e[m];
                    float* yp = y + (size_t)tok * DMODEL + n;
                    *yp += g * (acc[mt][nt][i] + b);
                }
            }
        }
}

// =====================================================================
// launch
// =====================================================================
extern "C" void kernel_launch(void* const* d_in, const int* in_sizes, int n_in,
                              void* d_out, int out_size, void* d_ws, size_t ws_size,
                              hipStream_t stream)
{
    (void)in_sizes; (void)n_in; (void)ws_size;
    const float* x     = (const float*)d_in[0];
    const float* wg    = (const float*)d_in[1];
    const float* fc1_w = (const float*)d_in[2];
    const float* fc1_b = (const float*)d_in[3];
    const float* fc2_w = (const float*)d_in[4];
    const float* fc2_b = (const float*)d_in[5];
    float* y = (float*)d_out;

    // workspace layout (~97 MB): disp bf16 | h bf16 (per-expert, reused) | small arrays
    char* ws = (char*)d_ws;
    __bf16* disp       = (__bf16*)ws;                                        // E*C*D bf16 = 64MB
    __bf16* hbuf       = (__bf16*)(ws + ((size_t)64 << 20));                 // C*H bf16   = 32MB
    int*    idxb       = (int*)  (ws + ((size_t)96 << 20));                  // S*2 int
    float*  gateb      = (float*)(ws + ((size_t)96 << 20) + (1u << 16));     // S*2 f32
    int*    slot_token = (int*)  (ws + ((size_t)96 << 20) + (2u << 16));     // E*C int
    float*  slot_gate  = (float*)(ws + ((size_t)96 << 20) + (3u << 16));     // E*C f32

    moe_gate_kernel<<<S_TOK / 8, 256, 0, stream>>>(x, wg, idxb, gateb);
    moe_scan_kernel<<<1, 256, 0, stream>>>(idxb, gateb, slot_token, slot_gate);
    moe_dispatch_kernel<<<NEXP * CAP, 256, 0, stream>>>(x, slot_token, disp);
    moe_zero_kernel<<<(out_size + 255) / 256, 256, 0, stream>>>(y, out_size);

    // per-expert FFN (serialized so one expert's 128MB weight slab lives in 192MB L2,
    // and so fc2's fused combine needs no atomics)
    for (int e = 0; e < NEXP; ++e) {
        const __bf16* disp_e = disp + (size_t)e * CAP * DMODEL;
        const float*  w1_e   = fc1_w + (size_t)e * DMODEL * HIDDEN;
        const float*  b1_e   = fc1_b + (size_t)e * HIDDEN;
        const float*  w2_e   = fc2_w + (size_t)e * HIDDEN * DMODEL;
        const float*  b2_e   = fc2_b + (size_t)e * DMODEL;

        dim3 g1(CAP / BM, HIDDEN / BN);   // 16 x 64
        moe_fc1_kernel<<<g1, 256, 0, stream>>>(disp_e, w1_e, b1_e, hbuf);

        dim3 g2(CAP / BM, DMODEL / BN);   // 16 x 16
        moe_fc2_combine_kernel<<<g2, 256, 0, stream>>>(
            hbuf, w2_e, b2_e, slot_token + (size_t)e * CAP,
            slot_gate + (size_t)e * CAP, y);
    }
}
